// ContactGraspnetModel_33543694581888
// MI455X (gfx1250) — matrix operand order repaired and checked by reading, not compiled, not run on hardware
//
#include <hip/hip_runtime.h>
#include <hip/hip_bf16.h>

typedef __attribute__((ext_vector_type(16))) _Float16 v16h;
typedef __attribute__((ext_vector_type(8)))  float    v8f;

#define NBATCH 4
#define NPTS   8192
#define NPOINT 1024
#define NSAMP  32
#define RAD2   0.01f
#define BNEPS  1e-5f

// ---------------------------------------------------------------------------
// Kernel 1: farthest point sampling.  One block (1024 threads) per batch.
// Each thread caches 8 points + running min-dist in registers; per-iteration
// work is a wave32 shuffle argmax + a tiny cross-wave LDS reduction.
// Emits new_xyz[b][i][3] for the selected centroids.
// ---------------------------------------------------------------------------
__global__ void fps_kernel(const float* __restrict__ xyz, float* __restrict__ new_xyz) {
    __shared__ float s_rv[32];
    __shared__ int   s_ri[32];
    __shared__ int   s_far;
    __shared__ float s_cx, s_cy, s_cz;

    const int b   = blockIdx.x;
    const int t   = threadIdx.x;
    const int ln  = t & 31;
    const int wid = t >> 5;
    const int b3N = b * 3 * NPTS;

    float px[8], py[8], pz[8], dist[8];
#pragma unroll
    for (int j = 0; j < 8; ++j) {
        int n = t + j * 1024;
        px[j] = xyz[b3N + 0 * NPTS + n];
        py[j] = xyz[b3N + 1 * NPTS + n];
        pz[j] = xyz[b3N + 2 * NPTS + n];
        dist[j] = 1e10f;
    }
    if (t == 0) s_far = 0;
    __syncthreads();

    for (int i = 0; i < NPOINT; ++i) {
        if (t == 0) {
            int f = s_far;
            float cx = xyz[b3N + 0 * NPTS + f];
            float cy = xyz[b3N + 1 * NPTS + f];
            float cz = xyz[b3N + 2 * NPTS + f];
            s_cx = cx; s_cy = cy; s_cz = cz;
            float* o = new_xyz + (b * NPOINT + i) * 3;
            o[0] = cx; o[1] = cy; o[2] = cz;
        }
        __syncthreads();
        float cx = s_cx, cy = s_cy, cz = s_cz;

        float bestv = -1.0f; int besti = 0;
#pragma unroll
        for (int j = 0; j < 8; ++j) {
            float dx = px[j] - cx, dy = py[j] - cy, dz = pz[j] - cz;
            float d = dx * dx + dy * dy + dz * dz;
            float dd = fminf(dist[j], d);
            dist[j] = dd;
            int n = t + j * 1024;
            if (dd > bestv || (dd == bestv && n < besti)) { bestv = dd; besti = n; }
        }
#pragma unroll
        for (int off = 16; off > 0; off >>= 1) {
            float ov = __shfl_down(bestv, off, 32);
            int   oi = __shfl_down(besti, off, 32);
            if (ov > bestv || (ov == bestv && oi < besti)) { bestv = ov; besti = oi; }
        }
        if (ln == 0) { s_rv[wid] = bestv; s_ri[wid] = besti; }
        __syncthreads();
        if (t < 32) {
            bestv = s_rv[t]; besti = s_ri[t];
#pragma unroll
            for (int off = 16; off > 0; off >>= 1) {
                float ov = __shfl_down(bestv, off, 32);
                int   oi = __shfl_down(besti, off, 32);
                if (ov > bestv || (ov == bestv && oi < besti)) { bestv = ov; besti = oi; }
            }
            if (t == 0) s_far = besti;
        }
        __syncthreads();
    }
}

// ---------------------------------------------------------------------------
// Kernel 2: ball query.  One wave32 per centroid; ballot + prefix-popc packs
// the first 32 in-radius indices (in ascending index order, matching the
// reference's sort-then-truncate), padding with the first hit.
// ---------------------------------------------------------------------------
__global__ void ball_query_kernel(const float* __restrict__ xyz,
                                  const float* __restrict__ new_xyz,
                                  int* __restrict__ ball_idx) {
    const int wave = (blockIdx.x * blockDim.x + threadIdx.x) >> 5;
    const int lane = threadIdx.x & 31;
    const int b    = wave / NPOINT;
    const int b3N  = b * 3 * NPTS;

    const float cx = new_xyz[wave * 3 + 0];
    const float cy = new_xyz[wave * 3 + 1];
    const float cz = new_xyz[wave * 3 + 2];
    int* out = ball_idx + wave * NSAMP;

    int cnt = 0;
    for (int n0 = 0; n0 < NPTS; n0 += 32) {
        int n = n0 + lane;
        float dx = xyz[b3N + 0 * NPTS + n] - cx;
        float dy = xyz[b3N + 1 * NPTS + n] - cy;
        float dz = xyz[b3N + 2 * NPTS + n] - cz;
        bool inb = (dx * dx + dy * dy + dz * dz) <= RAD2;
        unsigned mask = (unsigned)__ballot(inb);
        int pre  = __popc(mask & ((1u << lane) - 1u));
        int slot = cnt + pre;
        if (inb && slot < NSAMP) out[slot] = n;
        cnt += __popc(mask);
        if (cnt >= NSAMP) break;
    }
    if (cnt > NSAMP) cnt = NSAMP;
    if (cnt == 0) { if (lane == 0) out[0] = 0; cnt = 1; }
    int f = out[0];                       // same-wave store->load stays ordered
    for (int s = cnt + lane; s < NSAMP; s += 32) out[s] = f;
}

// ---------------------------------------------------------------------------
// Kernel 3: grouping + 3-layer MLP on v_wmma_f32_16x16x32_f16 + max-pool.
// Block = 256 threads = 8 waves = 4 centroids (2 waves / 32-sample group).
// Activations round-trip through a per-wave LDS slab in row-major [16][64]
// f16; A/B operands both use contiguous k = 16*half + s order.
// ---------------------------------------------------------------------------
__device__ inline v8f wmma16(v16h a, v16h b, v8f c) {
    return __builtin_amdgcn_wmma_f32_16x16x32_f16(false, a, false, b,
                                                  (short)0, c, false, false);
}

// A-tile (16x32 f16): lane row = lane&15; half = lane>>4 selects K ranges
// {0-15} (half 0) or {16-31} (half 1), 2 K's per VGPR.
__device__ inline v16h load_A(const _Float16* act, int row, int hf, int koff, int stride) {
    v16h a;
    const _Float16* p = act + row * stride + koff;
#pragma unroll
    for (int v = 0; v < 8; ++v) {
        int kb = 2 * v + 16 * hf;
        a[2 * v + 0] = p[kb + 0];
        a[2 * v + 1] = p[kb + 1];
    }
    return a;
}

// B-tile (32x16 f16) from row-major weights w[cout][cstride]:
// lane col n = lane&15 (+16*ntile applied by caller); element j -> K = j + 16*half.
__device__ inline v16h load_B(const _Float16* w, int n, int hf, int koff, int cstride) {
    v16h bv;
    const _Float16* p = w + n * cstride + koff + 16 * hf;
#pragma unroll
    for (int j = 0; j < 16; ++j) bv[j] = p[j];
    return bv;
}

__global__ void mlp_kernel(const float* __restrict__ xyz,
                           const float* __restrict__ points,
                           const float* __restrict__ new_xyz,
                           const int*   __restrict__ ball_idx,
                           const float* __restrict__ w1, const float* __restrict__ b1,
                           const float* __restrict__ g1, const float* __restrict__ t1,
                           const float* __restrict__ m1, const float* __restrict__ v1,
                           const float* __restrict__ w2, const float* __restrict__ b2,
                           const float* __restrict__ g2, const float* __restrict__ t2,
                           const float* __restrict__ m2, const float* __restrict__ v2,
                           const float* __restrict__ w3, const float* __restrict__ b3,
                           const float* __restrict__ g3, const float* __restrict__ t3,
                           const float* __restrict__ m3, const float* __restrict__ v3,
                           float* __restrict__ out) {
    __shared__ _Float16 s_w1[64 * 32];       // cin padded 6 -> 32 with zeros
    __shared__ _Float16 s_w2[64 * 64];
    __shared__ _Float16 s_w3[128 * 64];
    __shared__ float s_sc1[64], s_sh1[64];
    __shared__ float s_sc2[64], s_sh2[64];
    __shared__ float s_sc3[128], s_sh3[128];
    __shared__ _Float16 s_act[8][16 * 64];   // per-wave activation slab
    __shared__ float s_pmax[8][128];         // per-wave partial max

    const int tid = threadIdx.x;

    // ---- stage weights (f16) and folded BN scale/shift ----
    for (int i = tid; i < 64 * 32; i += 256) s_w1[i] = (_Float16)0.0f;
    __syncthreads();
    for (int i = tid; i < 64 * 6; i += 256)
        s_w1[(i / 6) * 32 + (i % 6)] = (_Float16)w1[i];
    for (int i = tid; i < 64 * 64; i += 256) s_w2[i] = (_Float16)w2[i];
    for (int i = tid; i < 128 * 64; i += 256) s_w3[i] = (_Float16)w3[i];
    for (int i = tid; i < 64; i += 256) {
        float sc = g1[i] * rsqrtf(v1[i] + BNEPS);
        s_sc1[i] = sc; s_sh1[i] = (b1[i] - m1[i]) * sc + t1[i];
        float sd = g2[i] * rsqrtf(v2[i] + BNEPS);
        s_sc2[i] = sd; s_sh2[i] = (b2[i] - m2[i]) * sd + t2[i];
    }
    for (int i = tid; i < 128; i += 256) {
        float sc = g3[i] * rsqrtf(v3[i] + BNEPS);
        s_sc3[i] = sc; s_sh3[i] = (b3[i] - m3[i]) * sc + t3[i];
    }
    __syncthreads();

    const int w     = tid >> 5;         // wave 0..7
    const int lane  = tid & 31;
    const int row   = lane & 15;
    const int hf    = lane >> 4;
    const int col   = lane & 15;
    const int mtile = w & 1;            // which half of the 32-sample group
    const int gg    = blockIdx.x * 4 + (w >> 1);   // global centroid
    const int bb    = gg >> 10;
    const int b3N   = bb * 3 * NPTS;

    _Float16* act = &s_act[w][0];

    // ---- build layer-1 input features into the slab (row-major 16x64) ----
    for (int i = lane; i < 16 * 64; i += 32) act[i] = (_Float16)0.0f;
    {
        int sample = mtile * 16 + row;
        int pidx = ball_idx[gg * NSAMP + sample];
        float nx = new_xyz[gg * 3 + 0];
        float ny = new_xyz[gg * 3 + 1];
        float nz = new_xyz[gg * 3 + 2];
        float fx = xyz[b3N + 0 * NPTS + pidx] - nx;
        float fy = xyz[b3N + 1 * NPTS + pidx] - ny;
        float fz = xyz[b3N + 2 * NPTS + pidx] - nz;
        float p0 = points[b3N + 0 * NPTS + pidx];
        float p1 = points[b3N + 1 * NPTS + pidx];
        float p2 = points[b3N + 2 * NPTS + pidx];
        if (hf == 0) {
            act[row * 64 + 0] = (_Float16)fx;
            act[row * 64 + 1] = (_Float16)fy;
            act[row * 64 + 2] = (_Float16)fz;
            act[row * 64 + 3] = (_Float16)p0;
            act[row * 64 + 4] = (_Float16)p1;
            act[row * 64 + 5] = (_Float16)p2;
        }
    }
    __syncthreads();

    // ---- layer 1: 16x6(->32) @ 32x64, BN+ReLU, back to slab ----
    {
        v16h a = load_A(act, row, hf, 0, 64);
#pragma unroll
        for (int nt = 0; nt < 4; ++nt) {
            v16h bw = load_B(s_w1, nt * 16 + col, hf, 0, 32);
            v8f c = {};
            c = wmma16(a, bw, c);
            int ch = nt * 16 + col;
            float sc = s_sc1[ch], sh = s_sh1[ch];
#pragma unroll
            for (int r = 0; r < 8; ++r) {
                float f = fmaxf(c[r] * sc + sh, 0.0f);
                act[(r + 8 * hf) * 64 + ch] = (_Float16)f;
            }
        }
    }
    __syncthreads();

    // ---- layer 2: 16x64 @ 64x64 (K chained), BN+ReLU, back to slab ----
    {
        v16h a0 = load_A(act, row, hf, 0, 64);
        v16h a1 = load_A(act, row, hf, 32, 64);
#pragma unroll
        for (int nt = 0; nt < 4; ++nt) {
            v16h bw0 = load_B(s_w2, nt * 16 + col, hf, 0, 64);
            v16h bw1 = load_B(s_w2, nt * 16 + col, hf, 32, 64);
            v8f c = {};
            c = wmma16(a0, bw0, c);
            c = wmma16(a1, bw1, c);
            int ch = nt * 16 + col;
            float sc = s_sc2[ch], sh = s_sh2[ch];
#pragma unroll
            for (int r = 0; r < 8; ++r) {
                float f = fmaxf(c[r] * sc + sh, 0.0f);
                act[(r + 8 * hf) * 64 + ch] = (_Float16)f;
            }
        }
    }
    __syncthreads();

    // ---- layer 3: 16x64 @ 64x128, BN+ReLU, in-register max over rows ----
    {
        v16h a0 = load_A(act, row, hf, 0, 64);
        v16h a1 = load_A(act, row, hf, 32, 64);
#pragma unroll
        for (int nt = 0; nt < 8; ++nt) {
            v16h bw0 = load_B(s_w3, nt * 16 + col, hf, 0, 64);
            v16h bw1 = load_B(s_w3, nt * 16 + col, hf, 32, 64);
            v8f c = {};
            c = wmma16(a0, bw0, c);
            c = wmma16(a1, bw1, c);
            int ch = nt * 16 + col;
            float sc = s_sc3[ch], sh = s_sh3[ch];
            float pm = -1e30f;
#pragma unroll
            for (int r = 0; r < 8; ++r) {
                float f = fmaxf(c[r] * sc + sh, 0.0f);
                pm = fmaxf(pm, f);
            }
            pm = fmaxf(pm, __shfl_xor(pm, 16, 32));  // combine row halves
            if (hf == 0) s_pmax[w][ch] = pm;         // max over this wave's 16 rows
        }
    }
    __syncthreads();

    // ---- final max across the 2 waves of each group; write (B,128,S) ----
    for (int i = tid; i < 4 * 128; i += 256) {
        int c  = i >> 7;
        int ch = i & 127;
        float mv = fmaxf(s_pmax[2 * c][ch], s_pmax[2 * c + 1][ch]);
        int g2i = blockIdx.x * 4 + c;
        int ob  = g2i >> 10;
        int os  = g2i & 1023;
        out[(ob * 128 + ch) * NPOINT + os] = mv;
    }
}

// ---------------------------------------------------------------------------
extern "C" void kernel_launch(void* const* d_in, const int* in_sizes, int n_in,
                              void* d_out, int out_size, void* d_ws, size_t ws_size,
                              hipStream_t stream) {
    const float* xyz    = (const float*)d_in[0];
    const float* points = (const float*)d_in[1];
    const float* w1 = (const float*)d_in[2];
    const float* b1 = (const float*)d_in[3];
    const float* g1 = (const float*)d_in[4];
    const float* t1 = (const float*)d_in[5];
    const float* m1 = (const float*)d_in[6];
    const float* v1 = (const float*)d_in[7];
    const float* w2 = (const float*)d_in[8];
    const float* b2 = (const float*)d_in[9];
    const float* g2 = (const float*)d_in[10];
    const float* t2 = (const float*)d_in[11];
    const float* m2 = (const float*)d_in[12];
    const float* v2 = (const float*)d_in[13];
    const float* w3 = (const float*)d_in[14];
    const float* b3 = (const float*)d_in[15];
    const float* g3 = (const float*)d_in[16];
    const float* t3 = (const float*)d_in[17];
    const float* m3 = (const float*)d_in[18];
    const float* v3 = (const float*)d_in[19];

    float* new_xyz = (float*)d_ws;                                   // 4*1024*3 f32
    int*   ball    = (int*)((char*)d_ws + NBATCH * NPOINT * 3 * 4);  // 4*1024*32 i32

    fps_kernel<<<NBATCH, 1024, 0, stream>>>(xyz, new_xyz);
    ball_query_kernel<<<(NBATCH * NPOINT) / 8, 256, 0, stream>>>(xyz, new_xyz, ball);
    mlp_kernel<<<(NBATCH * NPOINT) / 4, 256, 0, stream>>>(
        xyz, points, new_xyz, ball,
        w1, b1, g1, t1, m1, v1,
        w2, b2, g2, t2, m2, v2,
        w3, b3, g3, t3, m3, v3,
        (float*)d_out);
}
